// MoNE_24945170055223
// MI455X (gfx1250) — matrix operand order repaired and checked by reading, not compiled, not run on hardware
//
#include <hip/hip_runtime.h>
#include <hip/hip_bf16.h>
#include <math.h>

// ---------------------------------------------------------------------------
// MoNE ViT forward for MI455X (gfx1250, wave32, WMMA).
// Precision: f16 WMMA with f32 accumulation (V_WMMA_F32_16X16X32_F16).
// Attention: flash-style online softmax; K/V slabs double-buffered in LDS via
// GLOBAL_LOAD_ASYNC_TO_LDS_B64 (ASYNCcnt-tracked). GEMM A-tiles staged in LDS
// and shared by the 4 waves of a workgroup.
// ---------------------------------------------------------------------------

typedef _Float16 h16;
typedef __attribute__((ext_vector_type(16))) _Float16 v16h;
typedef __attribute__((ext_vector_type(8)))  float    v8f;
typedef __attribute__((ext_vector_type(2)))  int      v2i;

constexpr int kD    = 256;
constexpr int kE    = 4;
constexpr int kH    = 8;
constexpr int kL    = 8;
constexpr int kT    = 1024;
constexpr int kB    = 8;
constexpr int kNper = 256;     // tokens per expert block
constexpr int kDH   = 32;      // head dim
constexpr int kD4   = 1024;    // 4*D
constexpr int kCls  = 10;
constexpr int kImg  = 128;
constexpr float kScale = 0.0625f; // D^-0.5

// ----------------------- async LDS staging (guarded) -----------------------

#if __has_builtin(__builtin_amdgcn_global_load_async_to_lds_b64) && \
    __has_builtin(__builtin_amdgcn_s_wait_asynccnt)
#define MONE_ASYNC_LDS 1
#else
#define MONE_ASYNC_LDS 0
#endif

// Stage 8 bytes (4 halves) global -> LDS. Async (ASYNCcnt-tracked) when the
// gfx1250 builtins are available, else synchronous copy through VGPRs.
__device__ __forceinline__ void stage64(const h16* g, h16* l) {
#if MONE_ASYNC_LDS
  __builtin_amdgcn_global_load_async_to_lds_b64(
      (__attribute__((address_space(1))) v2i*)g,
      (__attribute__((address_space(3))) v2i*)l, 0, 0);
#else
  *(unsigned long long*)l = *(const unsigned long long*)g;
#endif
}

__device__ __forceinline__ void wait_async_all() {
#if MONE_ASYNC_LDS
  __builtin_amdgcn_s_wait_asynccnt(0);
#endif
}

// ------------------------------ WMMA helpers -------------------------------

__device__ __forceinline__ v8f wmma_f16(v16h a, v16h b, v8f c) {
  // D = A(16x32 f16) * B(32x16 f16) + C(16x16 f32)
  return __builtin_amdgcn_wmma_f32_16x16x32_f16(
      /*neg_a=*/false, a, /*neg_b=*/false, b,
      /*c_mod=*/(short)0, c, /*reuse_a=*/false, /*reuse_b=*/false);
}

// A-operand loader: 16(M) x 32(K) f16 tile, row stride lda (halves).
// Layout: lanes 0-15 rows M=0..15 hold K=0..15, lanes 16-31
// same rows hold K=16..31.
__device__ __forceinline__ v16h load_a16(const h16* base, int lda, int lane) {
  int m  = lane & 15;
  int kh = (lane >> 4) * 16;
  const h16* row = base + (size_t)m * lda;
  v16h a;
#pragma unroll
  for (int v = 0; v < 8; ++v) a[v]     = row[kh + v];
#pragma unroll
  for (int v = 0; v < 8; ++v) a[8 + v] = row[kh + 8 + v];
  return a;
}

// B-operand loader for OUT = X * W^T: B[k][n] = W[n][k], W row-major [N][K].
// Lane n (0-15) = column n holds K=0..15, lanes 16-31 hold K=16..31.
__device__ __forceinline__ v16h load_b_wt(const h16* w, int ldw, int lane) {
  int n    = lane & 15;
  int koff = (lane >> 4) * 16;
  const h16* row = w + (size_t)n * ldw + koff;
  v16h b;
#pragma unroll
  for (int i = 0; i < 16; ++i) b[i] = row[i];
  return b;
}

// B-operand loader for OUT = P * V: B[k][n] = V[k][n], V row-major [K][N].
__device__ __forceinline__ v16h load_b_cm(const h16* v, int ldv, int lane) {
  int n    = lane & 15;
  int koff = (lane >> 4) * 16;
  v16h b;
#pragma unroll
  for (int i = 0; i < 16; ++i) b[i] = v[(size_t)(koff + i) * ldv + n];
  return b;
}

__device__ __forceinline__ float gelu_f(float x) {
  return 0.5f * x * (1.0f + erff(x * 0.70710678118654752f));
}

// ------------------------------ small kernels ------------------------------

__global__ void f2h_kernel(const float* __restrict__ s, h16* __restrict__ d, int n) {
  int i = blockIdx.x * blockDim.x + threadIdx.x;
  if (i < n) d[i] = (h16)s[i];
}

// Patch embed: conv 4x4/4 as K=48 GEMV per (b,t,d) + bias + pos_emb -> tok f32
__global__ void patch_kernel(const float* __restrict__ x, const float* __restrict__ w,
                             const float* __restrict__ pb, const float* __restrict__ pos,
                             float* __restrict__ tok) {
  int d = threadIdx.x;              // 0..255
  int t = blockIdx.x;               // 0..1023
  int b = blockIdx.y;               // 0..7
  int pr = t >> 5, pc = t & 31;
  float acc = 0.f;
#pragma unroll
  for (int c = 0; c < 3; ++c)
#pragma unroll
    for (int i = 0; i < 4; ++i)
#pragma unroll
      for (int j = 0; j < 4; ++j)
        acc += x[(((size_t)b * 3 + c) * kImg + pr * 4 + i) * kImg + pc * 4 + j] *
               w[((d * 3 + c) * 4 + i) * 4 + j];
  tok[((size_t)b * kT + t) * kD + d] = acc + pb[d] + pos[(size_t)t * kD + d];
}

// Router: softmax over E=4 logits per token.
__global__ void router_kernel(const float* __restrict__ tok, const float* __restrict__ rw,
                              const float* __restrict__ rb, float* __restrict__ prob) {
  int gid = blockIdx.x * blockDim.x + threadIdx.x;     // b*T + t
  const float* row = tok + (size_t)gid * kD;
  float lg[kE];
  float mx = -3.0e38f;
#pragma unroll
  for (int e = 0; e < kE; ++e) {
    float a = rb[e];
    for (int k = 0; k < kD; ++k) a += row[k] * rw[e * kD + k];
    lg[e] = a; mx = fmaxf(mx, a);
  }
  float s = 0.f;
#pragma unroll
  for (int e = 0; e < kE; ++e) { lg[e] = __expf(lg[e] - mx); s += lg[e]; }
  float inv = 1.f / s;
#pragma unroll
  for (int e = 0; e < kE; ++e) prob[(size_t)gid * kE + e] = lg[e] * inv;
}

// Greedy masked top-k per expert via rank counting (ties broken by index).
__global__ void select_kernel(const float* __restrict__ prob, int* __restrict__ perm,
                              float* __restrict__ rp_sel) {
  __shared__ float pv[kT];
  __shared__ unsigned char taken[kT];
  int b = blockIdx.x, t = threadIdx.x;
  taken[t] = 0;
  __syncthreads();
  for (int e = 0; e < kE; ++e) {
    float mine = prob[((size_t)b * kT + t) * kE + e];
    pv[t] = taken[t] ? -1.f : mine;
    __syncthreads();
    float myv = pv[t];
    int rank = 0;
    for (int j = 0; j < kT; ++j) {
      float o = pv[j];
      rank += (o > myv) || (o == myv && j < t);
    }
    if (!taken[t] && rank < kNper) {
      perm[(size_t)b * kT + e * kNper + rank]   = t;
      rp_sel[(size_t)b * kT + e * kNper + rank] = mine;
      taken[t] = 1;
    }
    __syncthreads();
  }
}

__global__ void gather_kernel(const float* __restrict__ tok, const int* __restrict__ perm,
                              float* __restrict__ xs) {
  int d = threadIdx.x, slot = blockIdx.x, b = blockIdx.y;
  int src = perm[(size_t)b * kT + slot];
  xs[((size_t)b * kT + slot) * kD + d] = tok[((size_t)b * kT + src) * kD + d];
}

// LayerNorm f32 -> f16; one wave per token, 8 values per lane.
__global__ void ln_kernel(const float* __restrict__ xs, const float* __restrict__ g,
                          const float* __restrict__ bt, h16* __restrict__ out) {
  int token = blockIdx.x * 8 + (threadIdx.x >> 5);
  int lane  = threadIdx.x & 31;
  const float* row = xs + (size_t)token * kD;
  float v[8], s = 0.f;
#pragma unroll
  for (int i = 0; i < 8; ++i) { v[i] = row[lane + 32 * i]; s += v[i]; }
#pragma unroll
  for (int m = 16; m >= 1; m >>= 1) s += __shfl_xor(s, m, 32);
  float mu = s * (1.0f / kD);
  float q = 0.f;
#pragma unroll
  for (int i = 0; i < 8; ++i) { float d = v[i] - mu; q += d * d; }
#pragma unroll
  for (int m = 16; m >= 1; m >>= 1) q += __shfl_xor(q, m, 32);
  float inv = rsqrtf(q * (1.0f / kD) + 1e-5f);
#pragma unroll
  for (int i = 0; i < 8; ++i) {
    int c = lane + 32 * i;
    out[(size_t)token * kD + c] = (h16)((v[i] - mu) * inv * g[c] + bt[c]);
  }
}

// ------------------------ generic MoE WMMA GEMM ----------------------------
// OUT[b, e-block rows, :N] = act(A[rows,:K] @ W[:N,:K]^T + bias), optional
// in-place f32 residual accumulate (scaled by alpha*rp+1) and/or f16 output.
// The 16-row A tile is shared by all 4 waves of the block: staged in LDS,
// double-buffered, async.

struct GemmCfg {
  const h16*  A;      int lda;
  const h16*  W;      int ldw;
  const float* bias;
  float*      outF32;           // in-place accumulate target (xs), or null
  h16*        outF16;           // f16 output, or null
  int         out_ld;
  int         Nfix;             // 0 -> N = D>>e
  int         Kfix;             // 0 -> K = D>>e
  int         act;              // 0 none, 1 gelu
  int         scaled;           // 1: multiply by (alpha*rp+1) before accumulate
  const float* rp;
  const float* alpha;
};

__global__ void moe_gemm(GemmCfg c) {
  __shared__ h16 lds_a[2][16 * 40];              // 16x32 tile, stride 40
  int tid  = threadIdx.x;                        // 0..127 (4 waves)
  int lane = tid & 31;
  int wave = tid >> 5;
  int b = blockIdx.z >> 2;
  int e = blockIdx.z & 3;
  int m_width = kD >> e;
  int N = c.Nfix ? c.Nfix : m_width;
  int K = c.Kfix ? c.Kfix : m_width;
  int ntile  = blockIdx.x * 4 + wave;
  bool active = (ntile * 16) < N;
  int nt = active ? ntile : 0;                   // clamp reads, no writes
  int row0 = b * kT + e * kNper + blockIdx.y * 16;

  const h16* Abase = c.A + (size_t)row0 * c.lda;
  const h16* Wbase = c.W + (size_t)(nt * 16) * c.ldw;

  int sr = tid >> 3, sc = (tid & 7) * 4;         // 128 thr * 8B = 1KB tile
  stage64(Abase + (size_t)sr * c.lda + sc, &lds_a[0][sr * 40 + sc]);

  v8f acc = {0.f, 0.f, 0.f, 0.f, 0.f, 0.f, 0.f, 0.f};
  int nIter = K / 32;
  int cur = 0;
  for (int it = 0; it < nIter; ++it) {
    wait_async_all();
    __syncthreads();                              // lds_a[cur] ready, prev read done
    if (it + 1 < nIter)
      stage64(Abase + (it + 1) * 32 + (size_t)sr * c.lda + sc,
              &lds_a[cur ^ 1][sr * 40 + sc]);
    __builtin_prefetch(Wbase + it * 32 + 64, 0, 0);   // global_prefetch_b8
    v16h a  = load_a16(&lds_a[cur][0], 40, lane);     // ds_load
    v16h bw = load_b_wt(Wbase + it * 32, c.ldw, lane);
    acc = wmma_f16(a, bw, acc);
    cur ^= 1;
  }

  if (active) {
    int n  = lane & 15;
    int hi = lane >> 4;
    int gn = ntile * 16 + n;
    float bias = c.bias ? c.bias[gn] : 0.f;
#pragma unroll
    for (int r = 0; r < 8; ++r) {
      int gm  = row0 + r + 8 * hi;               // global row = b*T + token
      float v = acc[r] + bias;
      if (c.act == 1) v = gelu_f(v);
      if (c.outF16) c.outF16[(size_t)gm * c.out_ld + gn] = (h16)v;
      if (c.outF32) {
        float sf = 1.f;
        if (c.scaled) sf = c.alpha[0] * c.rp[gm] + 1.f;
        c.outF32[(size_t)gm * c.out_ld + gn] += sf * v;
      }
    }
  }
}

// ------------------------- flash attention kernel --------------------------
// One block = 8 waves sharing one (b, h); wave w handles q-tile qg*8+w.
// 32-key K/V slabs are staged cooperatively in LDS (double-buffered, async).

__global__ void attn_kernel(const h16* __restrict__ qf, const h16* __restrict__ kf,
                            const h16* __restrict__ vf, h16* __restrict__ aof) {
  __shared__ h16 ka[2][32 * 40];                 // 32 keys x 32 dims
  __shared__ h16 va[2][32 * 40];
  __shared__ h16 lds_p[8][16 * 32];              // per-wave P tile
  int tid  = threadIdx.x;
  int lane = tid & 31;
  int wave = tid >> 5;
  int b  = blockIdx.x >> 6;
  int rm = blockIdx.x & 63;
  int h  = rm >> 3;
  int qg = rm & 7;
  int qt = qg * 8 + wave;

  const h16* qbase = qf + ((size_t)(b * kT + qt * 16)) * kD + h * kDH;
  v16h aq = load_a16(qbase, kD, lane);

  v8f acc0 = {0,0,0,0,0,0,0,0}, acc1 = {0,0,0,0,0,0,0,0};
  v8f mrun, srun;
#pragma unroll
  for (int r = 0; r < 8; ++r) { mrun[r] = -3.0e38f; srun[r] = 0.f; }

  int col = lane & 15;
  int hi  = lane >> 4;
  h16* pt = &lds_p[wave][0];

  const h16* kroot = kf + (size_t)b * kT * kD + h * kDH;
  const h16* vroot = vf + (size_t)b * kT * kD + h * kDH;
  int sr = tid >> 3, sc = (tid & 7) * 4;         // 256 thr stage 32x32 halves

  stage64(kroot + (size_t)sr * kD + sc, &ka[0][sr * 40 + sc]);
  stage64(vroot + (size_t)sr * kD + sc, &va[0][sr * 40 + sc]);

  int cur = 0;
  for (int it = 0; it < kT / 32; ++it) {
    wait_async_all();
    __syncthreads();                              // slab[cur] ready
    if (it + 1 < kT / 32) {
      size_t base = (size_t)((it + 1) * 32 + sr) * kD + sc;
      stage64(kroot + base, &ka[cur ^ 1][sr * 40 + sc]);
      stage64(vroot + base, &va[cur ^ 1][sr * 40 + sc]);
    }

    v16h b0 = load_b_wt(&ka[cur][0],       40, lane);   // keys +0..15
    v16h b1 = load_b_wt(&ka[cur][16 * 40], 40, lane);   // keys +16..31
    v8f z = {0,0,0,0,0,0,0,0};
    v8f s0 = wmma_f16(aq, b0, z);
    v8f s1 = wmma_f16(aq, b1, z);

    v8f fac;
#pragma unroll
    for (int r = 0; r < 8; ++r) {
      float sc0 = s0[r] * kScale;
      float sc1 = s1[r] * kScale;
      float tm = fmaxf(sc0, sc1);
#pragma unroll
      for (int m = 1; m <= 8; m <<= 1) tm = fmaxf(tm, __shfl_xor(tm, m, 32));
      float mn   = fmaxf(mrun[r], tm);
      float corr = __expf(mrun[r] - mn);
      float p0   = __expf(sc0 - mn);
      float p1   = __expf(sc1 - mn);
      float rs   = p0 + p1;
#pragma unroll
      for (int m = 1; m <= 8; m <<= 1) rs += __shfl_xor(rs, m, 32);
      srun[r] = srun[r] * corr + rs;
      mrun[r] = mn;
      fac[r]  = corr;
      int mrow = r + 8 * hi;
      pt[mrow * 32 + col]      = (h16)p0;        // per-wave region, no barrier
      pt[mrow * 32 + 16 + col] = (h16)p1;
    }
#pragma unroll
    for (int r = 0; r < 8; ++r) { acc0[r] *= fac[r]; acc1[r] *= fac[r]; }

    v16h pa = load_a16(pt, 32, lane);            // P back as A-operand
    v16h v0 = load_b_cm(&va[cur][0],  40, lane); // dims 0..15
    v16h v1 = load_b_cm(&va[cur][16], 40, lane); // dims 16..31
    acc0 = wmma_f16(pa, v0, acc0);
    acc1 = wmma_f16(pa, v1, acc1);
    cur ^= 1;
  }

#pragma unroll
  for (int r = 0; r < 8; ++r) {
    float inv = 1.f / srun[r];
    size_t row = (size_t)(b * kT + qt * 16 + r + 8 * hi);
    aof[row * kD + h * kDH + col]      = (h16)(acc0[r] * inv);
    aof[row * kD + h * kDH + 16 + col] = (h16)(acc1[r] * inv);
  }
}

// ------------------------------- head --------------------------------------

__global__ void head_kernel(const float* __restrict__ xs, const float* __restrict__ hw,
                            const float* __restrict__ hb, float* __restrict__ out) {
  __shared__ float mv[kD];
  int b = blockIdx.x, d = threadIdx.x;
  float s = 0.f;
  for (int t = 0; t < kT; ++t) s += xs[((size_t)b * kT + t) * kD + d];
  mv[d] = s * (1.0f / kT);
  __syncthreads();
  if (d < kCls) {
    float a = hb[d];
    for (int k = 0; k < kD; ++k) a += mv[k] * hw[d * kD + k];
    out[b * kCls + d] = a;
  }
}

// ------------------------------ host side ----------------------------------

extern "C" void kernel_launch(void* const* d_in, const int* in_sizes, int n_in,
                              void* d_out, int out_size, void* d_ws, size_t ws_size,
                              hipStream_t stream) {
  (void)in_sizes; (void)n_in; (void)out_size; (void)ws_size;
  const float* x        = (const float*)d_in[0];
  const float* patch_w  = (const float*)d_in[1];
  const float* patch_b  = (const float*)d_in[2];
  const float* pos_emb  = (const float*)d_in[3];
  const float* router_w = (const float*)d_in[4];
  const float* router_b = (const float*)d_in[5];
  const float* ln1_g    = (const float*)d_in[6];
  const float* ln1_b    = (const float*)d_in[7];
  const float* q_w      = (const float*)d_in[8];
  const float* q_b      = (const float*)d_in[9];
  const float* k_w      = (const float*)d_in[10];
  const float* k_b      = (const float*)d_in[11];
  const float* v_w      = (const float*)d_in[12];
  const float* v_b      = (const float*)d_in[13];
  const float* o_w      = (const float*)d_in[14];
  const float* o_b      = (const float*)d_in[15];
  const float* ln2_g    = (const float*)d_in[16];
  const float* ln2_b    = (const float*)d_in[17];
  const float* l1_w     = (const float*)d_in[18];
  const float* l1_b     = (const float*)d_in[19];
  const float* l2_w     = (const float*)d_in[20];
  const float* l2_b     = (const float*)d_in[21];
  const float* head_w   = (const float*)d_in[22];
  const float* head_b   = (const float*)d_in[23];
  const float* alpha    = (const float*)d_in[24];
  float* out = (float*)d_out;

  // workspace carve-out
  char* w = (char*)d_ws;
  auto alloc = [&](size_t bytes) -> void* {
    void* p = (void*)w; w += (bytes + 255) & ~(size_t)255; return p;
  };
  float* tok    = (float*)alloc((size_t)kB * kT * kD * 4);
  float* xs     = (float*)alloc((size_t)kB * kT * kD * 4);
  h16*   xn     = (h16*)  alloc((size_t)kB * kT * kD * 2);
  h16*   qf     = (h16*)  alloc((size_t)kB * kT * kD * 2);
  h16*   kf     = (h16*)  alloc((size_t)kB * kT * kD * 2);
  h16*   vf     = (h16*)  alloc((size_t)kB * kT * kD * 2);
  h16*   aof    = (h16*)  alloc((size_t)kB * kT * kD * 2);
  h16*   inner  = (h16*)  alloc((size_t)kB * kT * kD4 * 2);
  float* prob   = (float*)alloc((size_t)kB * kT * kE * 4);
  float* rp_sel = (float*)alloc((size_t)kB * kT * 4);
  int*   perm   = (int*)  alloc((size_t)kB * kT * 4);
  h16*   wq  = (h16*)alloc((size_t)kL * kD * kD * 2);
  h16*   wk  = (h16*)alloc((size_t)kL * kD * kD * 2);
  h16*   wv  = (h16*)alloc((size_t)kL * kD * kD * 2);
  h16*   wo  = (h16*)alloc((size_t)kL * kD * kD * 2);
  h16*   wl1 = (h16*)alloc((size_t)kL * kD4 * kD * 2);
  h16*   wl2 = (h16*)alloc((size_t)kL * kD * kD4 * 2);

  auto cvt = [&](const float* s, h16* d, int n) {
    f2h_kernel<<<(n + 255) / 256, 256, 0, stream>>>(s, d, n);
  };
  cvt(q_w,  wq,  kL * kD * kD);
  cvt(k_w,  wk,  kL * kD * kD);
  cvt(v_w,  wv,  kL * kD * kD);
  cvt(o_w,  wo,  kL * kD * kD);
  cvt(l1_w, wl1, kL * kD4 * kD);
  cvt(l2_w, wl2, kL * kD * kD4);

  patch_kernel<<<dim3(kT, kB), kD, 0, stream>>>(x, patch_w, patch_b, pos_emb, tok);
  router_kernel<<<(kB * kT) / 256, 256, 0, stream>>>(tok, router_w, router_b, prob);
  select_kernel<<<kB, kT, 0, stream>>>(prob, perm, rp_sel);
  gather_kernel<<<dim3(kT, kB), 256, 0, stream>>>(tok, perm, xs);

  auto gemm = [&](const h16* A, int lda, const h16* W, int ldw, const float* bias,
                  float* o32, h16* o16, int old, int Nfix, int Kfix, int act,
                  int scaled, int maxNtiles) {
    GemmCfg c{A, lda, W, ldw, bias, o32, o16, old, Nfix, Kfix, act, scaled,
              rp_sel, alpha};
    dim3 g((maxNtiles + 3) / 4, kNper / 16, kB * kE);
    moe_gemm<<<g, 128, 0, stream>>>(c);
  };

  for (int l = 0; l < kL; ++l) {
    const h16* wq_l  = wq  + (size_t)l * kD * kD;
    const h16* wk_l  = wk  + (size_t)l * kD * kD;
    const h16* wv_l  = wv  + (size_t)l * kD * kD;
    const h16* wo_l  = wo  + (size_t)l * kD * kD;
    const h16* wl1_l = wl1 + (size_t)l * kD4 * kD;
    const h16* wl2_l = wl2 + (size_t)l * kD * kD4;

    ln_kernel<<<(kB * kT) / 8, 256, 0, stream>>>(xs, ln1_g + l * kD, ln1_b + l * kD, xn);

    gemm(xn, kD, wq_l, kD, q_b + l * kD, nullptr, qf, kD, kD, 0, 0, 0, kD / 16);
    gemm(xn, kD, wk_l, kD, k_b + l * kD, nullptr, kf, kD, kD, 0, 0, 0, kD / 16);
    gemm(xn, kD, wv_l, kD, v_b + l * kD, nullptr, vf, kD, kD, 0, 0, 0, kD / 16);

    attn_kernel<<<kB * kH * 8, 256, 0, stream>>>(qf, kf, vf, aof);

    // O projection, in-place residual: xs[:, :, :m] += ao[:, :, :m] @ o_w^T + b
    gemm(aof, kD, wo_l, kD, o_b + l * kD, xs, nullptr, kD, 0, 0, 0, 0, kD / 16);

    ln_kernel<<<(kB * kT) / 8, 256, 0, stream>>>(xs, ln2_g + l * kD, ln2_b + l * kD, xn);

    // MLP1: inner = gelu(xn[:, :, :m] @ l1_w[:, :m]^T + b), N = 4D
    gemm(xn, kD, wl1_l, kD, l1_b + l * kD4, nullptr, inner, kD4, kD4, 0, 1, 0, kD4 / 16);
    // MLP2: xs[:, :, :m] += (alpha*rp+1) * (inner @ l2_w[:m, :]^T + b)
    gemm(inner, kD4, wl2_l, kD4, l2_b + l * kD, xs, nullptr, kD, 0, kD4, 0, 1, kD / 16);
  }

  head_kernel<<<kB, kD, 0, stream>>>(xs, head_w, head_b, out);
}
